// MultiLabelRNN_1357209665654
// MI455X (gfx1250) — matrix operand order repaired and checked by reading, not compiled, not run on hardware
//
#include <hip/hip_runtime.h>
#include <hip/hip_bf16.h>

#define BATCH      1024
#define INPUTS_DIM 2048
#define HIDDEN     512
#define CLASSES    151
#define MAX_STEPS  65

typedef __attribute__((ext_vector_type(16))) __bf16 v16bf;
typedef __attribute__((ext_vector_type(8)))  float  v8f;
typedef __attribute__((ext_vector_type(4)))  unsigned int u32x4;
typedef __attribute__((ext_vector_type(8)))  int i32x8;
typedef __attribute__((ext_vector_type(4)))  int i32x4;

__device__ __forceinline__ unsigned short f2bf(float f) {
    unsigned int u = __float_as_uint(f);
    unsigned int r = u + 0x7fffu + ((u >> 16) & 1u);      // round-to-nearest-even
    if ((u & 0x7f800000u) == 0x7f800000u) r = u;          // inf / nan passthrough
    return (unsigned short)(r >> 16);
}

__device__ __forceinline__ float sigmoidf(float x) {
    return 1.0f / (1.0f + __expf(-x));
}

// ---------------------------------------------------------------------------
// Pack a row-major fp32 weight matrix W[N][K] (used as B in  C = A @ W^T)
// into bf16 fragments matching the CDNA5 32x16 (KxN) 16-bit B-matrix layout.
// Fragment (nTile, kc) = 32 lanes x 16 bf16 (1KB).  Lane: N = nTile*16+(l&15),
// half = l>>4 selects K group; element j -> v=j>>1, odd=j&1,
// K = kc*32 + 8*half + 2*v + odd + (v>=4 ? 8 : 0).
// ---------------------------------------------------------------------------
__global__ void pack_b_kernel(const float* __restrict__ W,
                              unsigned short* __restrict__ frag,
                              int N, int K) {
    size_t idx = (size_t)blockIdx.x * blockDim.x + threadIdx.x;
    size_t total = (size_t)N * K;
    if (idx >= total) return;
    int kcPerN   = K >> 5;                 // K/32 fragments along K
    size_t fragIdx = idx >> 9;             // / 512 elements per fragment
    int r    = (int)(idx & 511);
    int lane = r >> 4;
    int j    = r & 15;
    int nTile = (int)(fragIdx / kcPerN);
    int kc    = (int)(fragIdx % kcPerN);
    int n    = nTile * 16 + (lane & 15);
    int half = lane >> 4;
    int v    = j >> 1;
    int odd  = j & 1;
    int k    = kc * 32 + 8 * half + 2 * v + odd + ((v >= 4) ? 8 : 0);
    frag[idx] = f2bf(W[(size_t)n * K + k]);
}

__global__ void f32_to_bf16_kernel(const float* __restrict__ src,
                                   unsigned short* __restrict__ dst, int n) {
    int i = blockIdx.x * blockDim.x + threadIdx.x;
    if (i < n) dst[i] = f2bf(src[i]);
}

__global__ void zero_u32_kernel(unsigned int* __restrict__ p, int n) {
    int i = blockIdx.x * blockDim.x + threadIdx.x;
    if (i < n) p[i] = 0u;
}

// ---------------------------------------------------------------------------
// TDM helper: issue a 1-D tensor_load_to_lds moving `nElems` bf16 elements
// from `gaddr` (global) to LDS byte-offset `ldsOff`. D# per ISA 08 §8.3/8.4.
// ---------------------------------------------------------------------------
__device__ __forceinline__ void tdm_load_1d_bf16(unsigned long long gaddr,
                                                 unsigned int ldsOff,
                                                 unsigned int nElems) {
#if __has_builtin(__builtin_amdgcn_tensor_load_to_lds)
    // group 0: count=1 | lds_addr | global_addr[56:0] | type=2
    u32x4 g0;
    g0.x = 1u;                                       // count=1, user mode
    g0.y = ldsOff;                                   // lds_addr (bytes)
    g0.z = (unsigned int)(gaddr & 0xffffffffull);    // global_addr[31:0]
    g0.w = (unsigned int)((gaddr >> 32) & 0x1ffffffull) | (2u << 30); // [56:32] | type=2
    // group 1: wg_mask=0, data_size=1 (2B), tensor_dim0=tile_dim0=nElems (1-D)
    i32x8 g1;
    g1[0] = (int)(1u << 16);                         // data_size = 2 bytes
    g1[1] = (int)((nElems & 0xffffu) << 16);         // tensor_dim0[15:0] @ bits 63:48
    g1[2] = (int)(nElems >> 16);                     // tensor_dim0[31:16]; tensor_dim1=0
    g1[3] = (int)((nElems & 0xffffu) << 16);         // tile_dim0 @ bits 127:112
    g1[4] = 0;                                       // tile_dim1/2 unused
    g1[5] = (int)nElems;                             // tensor_dim0_stride[31:0]
    g1[6] = 0;
    g1[7] = 0;
    i32x4 z4 = {0, 0, 0, 0};
#if defined(__clang_major__) && (__clang_major__ >= 23)
    i32x8 z8 = {0, 0, 0, 0, 0, 0, 0, 0};
    __builtin_amdgcn_tensor_load_to_lds(g0, g1, z4, z4, z8, 0);
#else
    __builtin_amdgcn_tensor_load_to_lds(g0, g1, z4, z4, 0);
#endif
    __builtin_amdgcn_s_wait_tensorcnt(0);
#else
    (void)gaddr; (void)ldsOff; (void)nElems;
#endif
}

// ---------------------------------------------------------------------------
// bf16 WMMA GEMM:  C[M][N] = A[M][K] (bf16 row-major) x Bfrag + bias[N]
// Block = 8 waves = 256 threads; block tile = 16 (M) x 512 (N).
// Each wave owns a 16x64 strip: 4 fp32 accumulators; the A fragment is
// gathered once per K-chunk and reused by 4 v_wmma_f32_16x16x32_bf16.
// B fragments are explicitly double-buffered (kc+1 loads issued before kc's
// wmmas) and streamed ahead with unconditional speculative prefetch.
// A tile (16 x K bf16, contiguous) staged in LDS by the Tensor Data Mover
// (wave 0 issues tensor_load_to_lds, waits TENSORcnt, barrier releases all).
// ---------------------------------------------------------------------------
__global__ __launch_bounds__(256)
void gemm_bf16_wmma_kernel(const unsigned short* __restrict__ A,
                           const unsigned short* __restrict__ Bfrag,
                           const float* __restrict__ bias,
                           float* __restrict__ C,
                           unsigned short* __restrict__ Cbf,   // may be null
                           int M, int N, int K) {
    extern __shared__ unsigned short ldsA[];   // 16 * K bf16
    const int tid    = threadIdx.x;
    const int wave   = tid >> 5;
    const int lane   = tid & 31;
    const int mBase  = blockIdx.x * 16;
    const int nTile0 = (blockIdx.y * 8 + wave) * 4;   // first of 4 16-col tiles
    const int kcCount = K >> 5;

#if __has_builtin(__builtin_amdgcn_tensor_load_to_lds)
    if (wave == 0) {
        tdm_load_1d_bf16((unsigned long long)(uintptr_t)(A + (size_t)mBase * K),
                         (unsigned int)(uintptr_t)&ldsA[0],
                         (unsigned int)(16 * K));
    }
#else
    {   // cooperative fallback copy of the 16 x K A tile into LDS
        const unsigned int* g = reinterpret_cast<const unsigned int*>(A + (size_t)mBase * K);
        unsigned int* l = reinterpret_cast<unsigned int*>(ldsA);
        const int nwords = (16 * K) >> 1;
        for (int i = tid; i < nwords; i += 256) l[i] = g[i];
    }
#endif
    __syncthreads();

    v8f acc0 = {}, acc1 = {}, acc2 = {}, acc3 = {};
    const size_t fragStride = 512;  // elements per fragment
    const unsigned short* bp0 = Bfrag + ((size_t)(nTile0 + 0) * kcCount) * fragStride + lane * 16;
    const unsigned short* bp1 = Bfrag + ((size_t)(nTile0 + 1) * kcCount) * fragStride + lane * 16;
    const unsigned short* bp2 = Bfrag + ((size_t)(nTile0 + 2) * kcCount) * fragStride + lane * 16;
    const unsigned short* bp3 = Bfrag + ((size_t)(nTile0 + 3) * kcCount) * fragStride + lane * 16;

    // LDS row for this lane's A fragment: M = lane&15, base K offset 8*half
    const unsigned short* arow = ldsA + (size_t)(lane & 15) * K + ((lane >> 4) << 3);

    // prime the B double-buffer with kc = 0
    v16bf b0 = *reinterpret_cast<const v16bf*>(bp0);
    v16bf b1 = *reinterpret_cast<const v16bf*>(bp1);
    v16bf b2 = *reinterpret_cast<const v16bf*>(bp2);
    v16bf b3 = *reinterpret_cast<const v16bf*>(bp3);

    for (int kc = 0; kc < kcCount; ++kc) {
        // speculative stream-ahead; OOB prefetches are silently dropped
        __builtin_prefetch(bp0 + (size_t)(kc + 4) * fragStride, 0, 1);
        __builtin_prefetch(bp1 + (size_t)(kc + 4) * fragStride, 0, 1);
        __builtin_prefetch(bp2 + (size_t)(kc + 4) * fragStride, 0, 1);
        __builtin_prefetch(bp3 + (size_t)(kc + 4) * fragStride, 0, 1);

        // issue kc+1 B loads before kc's wmmas (last iter harmlessly reloads kc)
        const size_t knOff = (size_t)((kc + 1 < kcCount) ? kc + 1 : kc) * fragStride;
        v16bf n0 = *reinterpret_cast<const v16bf*>(bp0 + knOff);
        v16bf n1 = *reinterpret_cast<const v16bf*>(bp1 + knOff);
        v16bf n2 = *reinterpret_cast<const v16bf*>(bp2 + knOff);
        v16bf n3 = *reinterpret_cast<const v16bf*>(bp3 + knOff);

        // A fragment: ISA 16-bit A 16x32 layout, gathered as 8 x 32-bit pairs
        union { v16bf v; unsigned int u[8]; } af;
        const unsigned short* ar = arow + kc * 32;
#pragma unroll
        for (int v = 0; v < 8; ++v) {
            int koff = (v < 4) ? (2 * v) : (16 + 2 * (v - 4));
            af.u[v] = *reinterpret_cast<const unsigned int*>(ar + koff);
        }

        acc0 = __builtin_amdgcn_wmma_f32_16x16x32_bf16(false, af.v, false, b0, (short)0, acc0, false, false);
        acc1 = __builtin_amdgcn_wmma_f32_16x16x32_bf16(false, af.v, false, b1, (short)0, acc1, false, false);
        acc2 = __builtin_amdgcn_wmma_f32_16x16x32_bf16(false, af.v, false, b2, (short)0, acc2, false, false);
        acc3 = __builtin_amdgcn_wmma_f32_16x16x32_bf16(false, af.v, false, b3, (short)0, acc3, false, false);

        b0 = n0; b1 = n1; b2 = n2; b3 = n3;
    }

    // epilogue: C/D layout — lanes 0-15: N=lane, M=r ; lanes 16-31: N=lane-16, M=8+r
    const int mo = (lane >> 4) << 3;
    v8f accs[4] = {acc0, acc1, acc2, acc3};
#pragma unroll
    for (int i = 0; i < 4; ++i) {
        const int n  = (nTile0 + i) * 16 + (lane & 15);
        const float bb = bias ? bias[n] : 0.0f;
#pragma unroll
        for (int r = 0; r < 8; ++r) {
            const int m = mBase + mo + r;
            float val = accs[i][r] + bb;
            C[(size_t)m * N + n] = val;
            if (Cbf) Cbf[(size_t)m * N + n] = f2bf(val);
        }
    }
}

// ---------------------------------------------------------------------------
// Gates: LSTM + highway pointwise math. One thread per (b, h).
// ---------------------------------------------------------------------------
__global__ void gates_kernel(const float* __restrict__ pi,
                             const float* __restrict__ ps,
                             float* __restrict__ memF,
                             float* __restrict__ stateF,
                             unsigned short* __restrict__ state_bf,
                             float* __restrict__ outStates,
                             const int* __restrict__ obj_num, int t) {
    int idx = blockIdx.x * blockDim.x + threadIdx.x;
    if (idx >= BATCH * HIDDEN) return;
    int b = idx / HIDDEN;
    int h = idx - b * HIDDEN;
    const float* pib = pi + (size_t)b * (6 * HIDDEN);
    const float* psb = ps + (size_t)b * (5 * HIDDEN);

    float ig = sigmoidf(pib[h]              + psb[h]);
    float fg = sigmoidf(pib[HIDDEN + h]     + psb[HIDDEN + h]);
    float mi = tanhf   (pib[2 * HIDDEN + h] + psb[2 * HIDDEN + h]);
    float og = sigmoidf(pib[3 * HIDDEN + h] + psb[3 * HIDDEN + h]);
    float nm = ig * mi + fg * memF[idx];
    float out = og * tanhf(nm);
    float hg = sigmoidf(pib[4 * HIDDEN + h] + psb[4 * HIDDEN + h]);
    out = hg * out + (1.0f - hg) * pib[5 * HIDDEN + h];

    memF[idx]     = nm;
    stateF[idx]   = out;
    state_bf[idx] = f2bf(out);

    float maskv = (t < obj_num[b]) ? 1.0f : 0.0f;
    outStates[((size_t)b * MAX_STEPS + t) * HIDDEN + h] = out * maskv;
}

// ---------------------------------------------------------------------------
// Per-row head: logits -> softmax -> argmax(dist[1:]) -> tie-aware one-hot
// gather of W_onehot -> next embed (bf16). One block per batch row.
// ---------------------------------------------------------------------------
__global__ __launch_bounds__(256)
void row_kernel(const float* __restrict__ stateF,
                const float* __restrict__ W_out,  const float* __restrict__ b_out,
                const float* __restrict__ W_onehot, const float* __restrict__ b_onehot,
                const int* __restrict__ obj_num,
                float* __restrict__ outDists, float* __restrict__ outCommit,
                unsigned short* __restrict__ embed_bf, int t) {
    __shared__ float sd[CLASSES];
    __shared__ float sflag[CLASSES];
    __shared__ float red[2];
    __shared__ int   bestS;

    const int b   = blockIdx.x;
    const int tid = threadIdx.x;
    const float* x = stateF + (size_t)b * HIDDEN;

    if (tid < CLASSES) {
        float acc = b_out[tid];
        const float* w = W_out + (size_t)tid * HIDDEN;
        for (int k = 0; k < HIDDEN; ++k) acc += x[k] * w[k];
        sd[tid] = acc;
    }
    __syncthreads();
    if (tid == 0) {
        float mx = sd[0];
        for (int c = 1; c < CLASSES; ++c) mx = fmaxf(mx, sd[c]);
        red[0] = mx;
    }
    __syncthreads();
    if (tid < CLASSES) sd[tid] = __expf(sd[tid] - red[0]);
    __syncthreads();
    if (tid == 0) {
        float s = 0.0f;
        for (int c = 0; c < CLASSES; ++c) s += sd[c];
        red[1] = s;
    }
    __syncthreads();
    if (tid < CLASSES) sd[tid] *= (1.0f / red[1]);
    __syncthreads();
    if (tid == 0) {
        float mv = sd[1]; int bi = 1;
        for (int c = 2; c < CLASSES; ++c) { if (sd[c] > mv) { mv = sd[c]; bi = c; } }
        red[0] = mv; bestS = bi;
    }
    __syncthreads();

    const float mv    = red[0];
    const float maskv = (t < obj_num[b]) ? 1.0f : 0.0f;
    if (tid < CLASSES) {
        sflag[tid] = (sd[tid] == mv) ? 1.0f : 0.0f;
        outDists[((size_t)b * MAX_STEPS + t) * CLASSES + tid] = sd[tid] * maskv;
    }
    if (tid == 0) outCommit[(size_t)t * BATCH + b] = (float)(bestS - 1);
    __syncthreads();

    for (int h = tid; h < HIDDEN; h += blockDim.x) {
        float e = b_onehot[h];
        const float* w = W_onehot + (size_t)h * CLASSES;
        for (int c = 0; c < CLASSES; ++c) e += sflag[c] * w[c];
        embed_bf[(size_t)b * HIDDEN + h] = f2bf(e);
    }
}

// ---------------------------------------------------------------------------
extern "C" void kernel_launch(void* const* d_in, const int* in_sizes, int n_in,
                              void* d_out, int out_size, void* d_ws, size_t ws_size,
                              hipStream_t stream) {
    const float* inputs   = (const float*)d_in[0];
    const int*   obj_num  = (const int*)  d_in[1];
    const float* W_in     = (const float*)d_in[2];
    const float* b_in     = (const float*)d_in[3];
    const float* W_state  = (const float*)d_in[4];
    const float* b_state  = (const float*)d_in[5];
    const float* W_out    = (const float*)d_in[6];
    const float* b_out    = (const float*)d_in[7];
    const float* W_onehot = (const float*)d_in[8];
    const float* b_onehot = (const float*)d_in[9];
    const float* W_embed  = (const float*)d_in[10];
    const float* b_embed  = (const float*)d_in[11];

    float* out = (float*)d_out;
    const size_t distsN  = (size_t)BATCH * MAX_STEPS * CLASSES;
    const size_t commitN = (size_t)MAX_STEPS * BATCH;
    float* outDists  = out;
    float* outCommit = out + distsN;
    float* outStates = out + distsN + commitN;

    // workspace carve-up (256B aligned)
    char* w = (char*)d_ws;
    auto carve = [&](size_t bytes) { char* p = w; w += (bytes + 255) & ~(size_t)255; return p; };
    unsigned short* fragIn    = (unsigned short*)carve((size_t)6 * HIDDEN * HIDDEN * 2);       // 3072x512
    unsigned short* fragState = (unsigned short*)carve((size_t)5 * HIDDEN * HIDDEN * 2);       // 2560x512
    unsigned short* fragEmbed = (unsigned short*)carve((size_t)HIDDEN * INPUTS_DIM * 2);       // 512x2048
    unsigned short* inputs_bf = (unsigned short*)carve((size_t)BATCH * INPUTS_DIM * 2);
    unsigned short* embed_bf  = (unsigned short*)carve((size_t)BATCH * HIDDEN * 2);
    unsigned short* state_bf  = (unsigned short*)carve((size_t)BATCH * HIDDEN * 2);
    float* embed_f = (float*)carve((size_t)BATCH * HIDDEN * 4);
    float* state_f = (float*)carve((size_t)BATCH * HIDDEN * 4);
    float* mem_f   = (float*)carve((size_t)BATCH * HIDDEN * 4);
    float* pi_buf  = (float*)carve((size_t)BATCH * 6 * HIDDEN * 4);
    float* ps_buf  = (float*)carve((size_t)BATCH * 5 * HIDDEN * 4);
    (void)ws_size; (void)n_in; (void)in_sizes; (void)out_size;

    // ---- prep: pack weights to bf16 fragments, convert inputs, zero state ----
    {
        int tot;
        tot = 6 * HIDDEN * HIDDEN;
        pack_b_kernel<<<(tot + 255) / 256, 256, 0, stream>>>(W_in, fragIn, 6 * HIDDEN, HIDDEN);
        tot = 5 * HIDDEN * HIDDEN;
        pack_b_kernel<<<(tot + 255) / 256, 256, 0, stream>>>(W_state, fragState, 5 * HIDDEN, HIDDEN);
        tot = HIDDEN * INPUTS_DIM;
        pack_b_kernel<<<(tot + 255) / 256, 256, 0, stream>>>(W_embed, fragEmbed, HIDDEN, INPUTS_DIM);
        tot = BATCH * INPUTS_DIM;
        f32_to_bf16_kernel<<<(tot + 255) / 256, 256, 0, stream>>>(inputs, inputs_bf, tot);
        int z16 = (BATCH * HIDDEN) / 2;  // ushorts as u32
        zero_u32_kernel<<<(z16 + 255) / 256, 256, 0, stream>>>((unsigned int*)state_bf, z16);
        int z32 = BATCH * HIDDEN;
        zero_u32_kernel<<<(z32 + 255) / 256, 256, 0, stream>>>((unsigned int*)mem_f, z32);
    }

    // ---- embed0 = inputs @ W_embed^T + b_embed  (also emit bf16 copy) ----
    {
        dim3 grid(BATCH / 16, HIDDEN / 512);
        size_t lds = (size_t)16 * INPUTS_DIM * 2;   // 64 KB
        gemm_bf16_wmma_kernel<<<grid, 256, lds, stream>>>(
            inputs_bf, fragEmbed, b_embed, embed_f, embed_bf,
            BATCH, HIDDEN, INPUTS_DIM);
    }

    // ---- 65 recurrent steps ----
    const size_t ldsStep = (size_t)16 * HIDDEN * 2;  // 16 KB
    for (int t = 0; t < MAX_STEPS; ++t) {
        {
            dim3 grid(BATCH / 16, (6 * HIDDEN) / 512);
            gemm_bf16_wmma_kernel<<<grid, 256, ldsStep, stream>>>(
                embed_bf, fragIn, b_in, pi_buf, (unsigned short*)nullptr,
                BATCH, 6 * HIDDEN, HIDDEN);
        }
        {
            dim3 grid(BATCH / 16, (5 * HIDDEN) / 512);
            gemm_bf16_wmma_kernel<<<grid, 256, ldsStep, stream>>>(
                state_bf, fragState, b_state, ps_buf, (unsigned short*)nullptr,
                BATCH, 5 * HIDDEN, HIDDEN);
        }
        {
            int tot = BATCH * HIDDEN;
            gates_kernel<<<(tot + 255) / 256, 256, 0, stream>>>(
                pi_buf, ps_buf, mem_f, state_f, state_bf, outStates, obj_num, t);
        }
        {
            row_kernel<<<BATCH, 256, 0, stream>>>(
                state_f, W_out, b_out, W_onehot, b_onehot, obj_num,
                outDists, outCommit, embed_bf, t);
        }
    }
}
